// ARDConv_71098888618710
// MI455X (gfx1250) — matrix operand order repaired and checked by reading, not compiled, not run on hardware
//
#include <hip/hip_runtime.h>
#include <hip/hip_bf16.h>
#include <math.h>

// ---------------- CDNA5 WMMA types ----------------
typedef _Float16 h8   __attribute__((ext_vector_type(8)));
typedef _Float16 v16h __attribute__((ext_vector_type(16)));
typedef float    v8f  __attribute__((ext_vector_type(8)));

__device__ inline v16h mk16(h8 a, h8 b) {
  v16h r;
#pragma unroll
  for (int i = 0; i < 8; ++i) { r[i] = a[i]; r[i + 8] = b[i]; }
  return r;
}
__device__ inline v8f wmma_f16(v16h a, v16h b, v8f c) {
  return __builtin_amdgcn_wmma_f32_16x16x32_f16(false, a, false, b, (short)0, c, false, false);
}
__device__ inline v8f zero8() {
  v8f z;
#pragma unroll
  for (int i = 0; i < 8; ++i) z[i] = 0.f;
  return z;
}
__device__ inline float sigm(float v) { return 1.f / (1.f + __expf(-v)); }
__device__ inline float silu(float v) { return v * sigm(v); }

// Problem constants
// B=16 C=128 H=W=64 L=4096 O=128 K=4 HID=32 F=C*9=1152 NSET=5

// ---------------- mean over L (fp32 / f16 input) ----------------
__global__ __launch_bounds__(256) void reduce_mean_f32(const float* __restrict__ in, float* __restrict__ out) {
  __shared__ float red[256];
  int bc = blockIdx.x, tid = threadIdx.x;
  float s = 0.f;
  for (int i = tid; i < 4096; i += 256) s += in[((size_t)bc << 12) + i];
  red[tid] = s; __syncthreads();
  for (int off = 128; off > 0; off >>= 1) { if (tid < off) red[tid] += red[tid + off]; __syncthreads(); }
  if (tid == 0) out[bc] = red[0] * (1.f / 4096.f);
}
__global__ __launch_bounds__(256) void reduce_mean_f16(const _Float16* __restrict__ in, float* __restrict__ out) {
  __shared__ float red[256];
  int bc = blockIdx.x, tid = threadIdx.x;
  float s = 0.f;
  for (int i = tid; i < 4096; i += 256) s += (float)in[((size_t)bc << 12) + i];
  red[tid] = s; __syncthreads();
  for (int off = 128; off > 0; off >>= 1) { if (tid < off) red[tid] += red[tid + off]; __syncthreads(); }
  if (tid == 0) out[bc] = red[0] * (1.f / 4096.f);
}

// ---------------- global-context + channel gates (tiny MLPs) ----------------
__global__ __launch_bounds__(256) void gates_kernel(
    const float* __restrict__ p, const float* __restrict__ gc_w1, const float* __restrict__ gc_w2,
    const float* __restrict__ cg_w1, const float* __restrict__ cg_b1,
    const float* __restrict__ cg_w2, const float* __restrict__ cg_b2,
    float* __restrict__ g_out, float* __restrict__ ca_out) {
  __shared__ float sp[2048];   // p, then pc in-place
  __shared__ float sh1[512];
  __shared__ float sh2[256];
  int tid = threadIdx.x;
  for (int i = tid; i < 2048; i += 256) sp[i] = p[i];
  __syncthreads();
  for (int i = tid; i < 512; i += 256) {           // silu(p @ gc_w1^T)  [16,32]
    int b = i >> 5, h = i & 31; float s = 0.f;
    for (int c = 0; c < 128; ++c) s += sp[b * 128 + c] * gc_w1[h * 128 + c];
    sh1[i] = silu(s);
  }
  __syncthreads();
  for (int i = tid; i < 2048; i += 256) {          // g = sigmoid(. @ gc_w2^T); pc = p*g
    int b = i >> 7, c = i & 127; float s = 0.f;
    for (int h = 0; h < 32; ++h) s += sh1[b * 32 + h] * gc_w2[c * 32 + h];
    float gg = sigm(s);
    g_out[i] = gg; sp[i] *= gg;
  }
  __syncthreads();
  for (int i = tid; i < 256; i += 256) {           // silu(pc @ cg_w1^T + b1)  [16,16]
    int b = i >> 4, j = i & 15; float s = cg_b1[j];
    for (int c = 0; c < 128; ++c) s += sp[b * 128 + c] * cg_w1[j * 128 + c];
    sh2[i] = silu(s);
  }
  __syncthreads();
  for (int i = tid; i < 2048; i += 256) {          // c_attn = sigmoid(. @ cg_w2^T + b2)
    int b = i >> 7, c = i & 127; float s = cg_b2[c];
    for (int j = 0; j < 16; ++j) s += sh2[b * 16 + j] * cg_w2[c * 16 + j];
    ca_out[i] = sigm(s);
  }
}

// ---------------- spatial gate: channel mean/max of x*g ----------------
__global__ __launch_bounds__(256) void spatial_stats(const float* __restrict__ x, const float* __restrict__ g,
                                                     float* __restrict__ sin2) {
  __shared__ float sg[128];
  int n = blockIdx.x * 256 + threadIdx.x;          // < 65536, block stays in one b
  int b = n >> 12, l = n & 4095;
  if (threadIdx.x < 128) sg[threadIdx.x] = g[b * 128 + threadIdx.x];
  __syncthreads();
  float mean = 0.f, mx = -1e30f;
  for (int c = 0; c < 128; ++c) {
    float v = x[(((size_t)b * 128 + c) << 12) + l] * sg[c];
    mean += v; mx = fmaxf(mx, v);
  }
  sin2[(((size_t)b * 2) << 12) + l]     = mean * (1.f / 128.f);
  sin2[(((size_t)b * 2 + 1) << 12) + l] = mx;
}

// ---------------- 7x7 spatial-gate conv, pad 3 ----------------
__global__ __launch_bounds__(256) void sg_conv(const float* __restrict__ sin2, const float* __restrict__ sgw,
                                               float* __restrict__ sattn) {
  __shared__ float sw[98];
  if (threadIdx.x < 98) sw[threadIdx.x] = sgw[threadIdx.x];
  __syncthreads();
  int n = blockIdx.x * 256 + threadIdx.x;
  int b = n >> 12, l = n & 4095, h = l >> 6, w = l & 63;
  float s = 0.f;
  for (int ch = 0; ch < 2; ++ch)
    for (int i = 0; i < 7; ++i) {
      int hh = h + i - 3; if (hh < 0 || hh > 63) continue;
      for (int j = 0; j < 7; ++j) {
        int ww = w + j - 3; if (ww < 0 || ww > 63) continue;
        s += sin2[(((size_t)(b * 2 + ch)) << 12) + (hh << 6) + ww] * sw[ch * 49 + i * 7 + j];
      }
    }
  sattn[n] = sigm(s);
}

// ---------------- x2 = x*g*c_attn*s_attn  (f16 for WMMA) ----------------
__global__ __launch_bounds__(256) void x2_kernel(const float* __restrict__ x, const float* __restrict__ g,
                                                 const float* __restrict__ ca, const float* __restrict__ sattn,
                                                 _Float16* __restrict__ x2h) {
  size_t n = (size_t)blockIdx.x * 256 + threadIdx.x;  // < 8388608
  int bc = (int)(n >> 12), b = bc >> 7, l = (int)(n & 4095);
  float v = x[n] * g[bc] * ca[bc] * sattn[(b << 12) + l];
  x2h[n] = (_Float16)v;
}

// ---------------- gf = silu(mean(x2) @ rg_w^T) ----------------
__global__ __launch_bounds__(256) void gf_kernel(const float* __restrict__ px2, const float* __restrict__ rgw,
                                                 float* __restrict__ gf) {
  for (int i = threadIdx.x; i < 512; i += 256) {
    int b = i >> 5, h = i & 31; float s = 0.f;
    for (int c = 0; c < 128; ++c) s += px2[b * 128 + c] * rgw[h * 128 + c];
    gf[i] = silu(s);
  }
}

// ---------------- f16 weight prep: wcat[640,1152] = {base, delta_k*scale}; rc_h; gate_h ----------------
__global__ __launch_bounds__(256) void prep_weights(const float* __restrict__ base, const float* __restrict__ delta,
                                                    const float* __restrict__ ksc, const float* __restrict__ rcw,
                                                    const float* __restrict__ gtw,
                                                    _Float16* __restrict__ wcat, _Float16* __restrict__ rch,
                                                    _Float16* __restrict__ gth) {
  int i = blockIdx.x * 256 + threadIdx.x;
  if (i < 737280) {
    int nr = i / 1152, f = i - nr * 1152; float v;
    if (nr < 128) v = base[nr * 1152 + f];
    else { int m = nr - 128, k = m >> 7, o = m & 127;
           v = delta[(size_t)(k * 128 + o) * 1152 + f] * ksc[k * 128 + o]; }
    wcat[i] = (_Float16)v;
  } else if (i < 737280 + 36864) {
    int j = i - 737280; rch[j] = (_Float16)rcw[j];
  } else if (i < 737280 + 36864 + 16384) {
    int j = i - 737280 - 36864; gth[j] = (_Float16)gtw[j];
  }
}

__global__ __launch_bounds__(256) void zero_f32(float* __restrict__ p, int n) {
  int i = blockIdx.x * 256 + threadIdx.x; if (i < n) p[i] = 0.f;
}

// ---------------- router: dilated 3x3 conv (implicit GEMM, WMMA) + BN + SiLU + gf ----------------
// grid (L/16, B), 64 threads = 2 waves, wave w owns o-tile w (N=32 total)
__global__ __launch_bounds__(64) void router_wmma(const _Float16* __restrict__ x2h, const _Float16* __restrict__ rch,
                                                  const float* __restrict__ bng, const float* __restrict__ bnb,
                                                  const float* __restrict__ bnm, const float* __restrict__ bnv,
                                                  const float* __restrict__ gf, _Float16* __restrict__ lfh) {
  __shared__ __align__(16) _Float16 As[16 * 1152];   // 36 KB im2col tile (dilation 2, pad 2)
  const int b = blockIdx.y, mbase = blockIdx.x * 16;
  const int h0 = mbase >> 6, w0 = mbase & 63;
  const int tid = threadIdx.x;
  for (int idx = tid; idx < 16 * 1152; idx += 64) {
    int m = idx / 1152, f = idx - m * 1152;
    int c = f / 9, t = f - c * 9;
    int hh = h0 + 2 * (t / 3) - 2, ww = w0 + m + 2 * (t % 3) - 2;
    _Float16 v = (_Float16)0.f;
    if (hh >= 0 && hh < 64 && ww >= 0 && ww < 64)
      v = x2h[(((size_t)(b * 128 + c)) << 12) + (hh << 6) + ww];
    As[idx] = v;
  }
  __syncthreads();
  const int wv = tid >> 5, lane = tid & 31, l2 = lane & 15, hi = lane >> 4;
  v8f acc = zero8();
  for (int ks = 0; ks < 36; ++ks) {
    const h8* ap = (const h8*)(As + l2 * 1152 + ks * 32 + hi * 8);
    v16h av = mk16(ap[0], ap[2]);                             // A: K {hi*8..}+{16+hi*8..}
    const h8* bp = (const h8*)(rch + (size_t)(wv * 16 + l2) * 1152 + ks * 32 + hi * 8);
    v16h bv = mk16(bp[0], bp[2]);                             // B: K {hi*8..}+{16+hi*8..}
    acc = wmma_f16(av, bv, acc);
  }
  const int hch = wv * 16 + l2;
  float rs = rsqrtf(bnv[hch] + 1e-5f), ga = bng[hch], be = bnb[hch], mu = bnm[hch];
  float gfv = gf[b * 32 + hch];
#pragma unroll
  for (int r = 0; r < 8; ++r) {
    int l = mbase + r + 8 * hi;
    float v = (acc[r] - mu) * rs * ga + be;
    v = silu(v) + gfv;
    lfh[(((size_t)(b * 4096 + l)) << 5) + hch] = (_Float16)v;
  }
}

// ---------------- main fused dynamic conv: 5-set implicit GEMM + WMMA alpha gate + softmax combine ----------------
// grid (L/16, B), 256 threads = 8 waves; wave w owns o-tile w (16 channels), all 5 weight sets
__global__ __launch_bounds__(256) void main_wmma(const _Float16* __restrict__ x2h, const _Float16* __restrict__ wcat,
                                                 const _Float16* __restrict__ lfh, const _Float16* __restrict__ gth,
                                                 const float* __restrict__ gate_b, float* __restrict__ outg,
                                                 float* __restrict__ sums, float* __restrict__ sumsq) {
  __shared__ __align__(16) _Float16 As[16 * 1152];   // 36 KB im2col tile (3x3, pad 1)
  const int b = blockIdx.y, mbase = blockIdx.x * 16;
  const int h0 = mbase >> 6, w0 = mbase & 63;
  const int tid = threadIdx.x;
  for (int idx = tid; idx < 16 * 1152; idx += 256) {
    int m = idx / 1152, f = idx - m * 1152;
    int c = f / 9, t = f - c * 9;
    int hh = h0 + (t / 3) - 1, ww = w0 + m + (t % 3) - 1;
    _Float16 v = (_Float16)0.f;
    if (hh >= 0 && hh < 64 && ww >= 0 && ww < 64)
      v = x2h[(((size_t)(b * 128 + c)) << 12) + (hh << 6) + ww];
    As[idx] = v;
  }
  __syncthreads();
  const int wv = tid >> 5, lane = tid & 31, l2 = lane & 15, hi = lane >> 4;

  v8f acc[5];
#pragma unroll
  for (int s = 0; s < 5; ++s) acc[s] = zero8();

  for (int ks = 0; ks < 36; ++ks) {                            // K = 1152 = 36 x 32
    const h8* ap = (const h8*)(As + l2 * 1152 + ks * 32 + hi * 8);
    v16h av = mk16(ap[0], ap[2]);
#pragma unroll
    for (int s = 0; s < 5; ++s) {
      const h8* bp = (const h8*)(wcat + (size_t)(s * 128 + wv * 16 + l2) * 1152 + ks * 32 + hi * 8);
      v16h bv = mk16(bp[0], bp[2]);
      acc[s] = wmma_f16(av, bv, acc[s]);
    }
  }

  // alpha logits via WMMA: lf[16 x 32] x gate_w^T — C-layout matches acc tiles exactly
  v8f lg[4];
#pragma unroll
  for (int k = 0; k < 4; ++k) lg[k] = zero8();
  {
    const h8* ap = (const h8*)(lfh + (((size_t)(b * 4096 + mbase + l2)) << 5) + hi * 8);
    v16h av = mk16(ap[0], ap[2]);
#pragma unroll
    for (int k = 0; k < 4; ++k) {
      const h8* bp = (const h8*)(gth + ((size_t)(k * 128 + wv * 16 + l2) << 5) + hi * 8);
      v16h bv = mk16(bp[0], bp[2]);
      lg[k] = wmma_f16(av, bv, lg[k]);
    }
  }

  const int o = wv * 16 + l2;
  float gb0 = gate_b[o], gb1 = gate_b[128 + o], gb2 = gate_b[256 + o], gb3 = gate_b[384 + o];
  const float itau = 1.f / 1.5f;
  float psum = 0.f, psq = 0.f;
#pragma unroll
  for (int r = 0; r < 8; ++r) {
    float z0 = (lg[0][r] + gb0) * itau, z1 = (lg[1][r] + gb1) * itau;
    float z2 = (lg[2][r] + gb2) * itau, z3 = (lg[3][r] + gb3) * itau;
    float mx = fmaxf(fmaxf(z0, z1), fmaxf(z2, z3));
    float e0 = __expf(z0 - mx), e1 = __expf(z1 - mx), e2 = __expf(z2 - mx), e3 = __expf(z3 - mx);
    float inv = 1.f / (e0 + e1 + e2 + e3);
    float val = acc[0][r] + (e0 * acc[1][r] + e1 * acc[2][r] + e2 * acc[3][r] + e3 * acc[4][r]) * inv;
    int l = mbase + r + 8 * hi;
    outg[(((size_t)(b * 128 + o)) << 12) + l] = val;
    psum += val; psq += val * val;
  }
  atomicAdd(&sums[b * 128 + o], psum);
  atomicAdd(&sumsq[b * 128 + o], psq);
}

// ---------------- GroupNorm stats + SE -> per-(b,o) affine ----------------
__global__ __launch_bounds__(256) void gn_se(const float* __restrict__ sums, const float* __restrict__ sumsq,
                                             const float* __restrict__ gng, const float* __restrict__ gnb,
                                             const float* __restrict__ sw1, const float* __restrict__ sb1,
                                             const float* __restrict__ sw2, const float* __restrict__ sb2,
                                             const float* __restrict__ bias,
                                             float* __restrict__ scl, float* __restrict__ shf) {
  __shared__ float smu[128], srs[128];   // [b][group]
  __shared__ float sps[2048];            // per-channel post-GN spatial mean
  __shared__ float sh[256];
  int tid = threadIdx.x;
  for (int i = tid; i < 128; i += 256) {
    int b = i >> 3, gi = i & 7; float s = 0.f, q = 0.f;
    for (int j = 0; j < 16; ++j) { int o = gi * 16 + j; s += sums[b * 128 + o]; q += sumsq[b * 128 + o]; }
    float n = 16.f * 4096.f, mu = s / n, var = q / n - mu * mu;
    smu[i] = mu; srs[i] = rsqrtf(var + 1e-5f);
  }
  __syncthreads();
  for (int i = tid; i < 2048; i += 256) {
    int b = i >> 7, o = i & 127, gi = o >> 4;
    float mean_o = sums[i] * (1.f / 4096.f);
    sps[i] = (mean_o - smu[b * 8 + gi]) * srs[b * 8 + gi] * gng[o] + gnb[o];
  }
  __syncthreads();
  for (int i = tid; i < 256; i += 256) {
    int b = i >> 4, j = i & 15; float s = sb1[j];
    for (int o = 0; o < 128; ++o) s += sps[b * 128 + o] * sw1[j * 128 + o];
    sh[i] = silu(s);
  }
  __syncthreads();
  for (int i = tid; i < 2048; i += 256) {
    int b = i >> 7, o = i & 127, gi = o >> 4; float s = sb2[o];
    for (int j = 0; j < 16; ++j) s += sh[b * 16 + j] * sw2[o * 16 + j];
    float se = sigm(s);
    float a = srs[b * 8 + gi] * gng[o] * se;
    scl[i] = a;
    shf[i] = (gnb[o] - smu[b * 8 + gi] * srs[b * 8 + gi] * gng[o]) * se + bias[o];
  }
}

// ---------------- final: affine + residual + bias ----------------
__global__ __launch_bounds__(256) void final_kernel(const float* __restrict__ outg, const float* __restrict__ x,
                                                    const float* __restrict__ scl, const float* __restrict__ shf,
                                                    const float* __restrict__ idsc, float* __restrict__ out) {
  size_t n = (size_t)blockIdx.x * 256 + threadIdx.x;
  int bo = (int)(n >> 12);
  out[n] = outg[n] * scl[bo] + shf[bo] + tanhf(idsc[0]) * x[n];
}

// =======================================================================
extern "C" void kernel_launch(void* const* d_in, const int* in_sizes, int n_in,
                              void* d_out, int out_size, void* d_ws, size_t ws_size,
                              hipStream_t stream) {
  const float* x        = (const float*)d_in[0];
  const float* base_w   = (const float*)d_in[1];
  const float* delta_w  = (const float*)d_in[2];
  const float* ksc      = (const float*)d_in[3];
  const float* gc_w1    = (const float*)d_in[4];
  const float* gc_w2    = (const float*)d_in[5];
  const float* rc_w     = (const float*)d_in[6];
  const float* bn_g     = (const float*)d_in[7];
  const float* bn_b     = (const float*)d_in[8];
  const float* bn_m     = (const float*)d_in[9];
  const float* bn_v     = (const float*)d_in[10];
  const float* rg_w     = (const float*)d_in[11];
  const float* gate_w   = (const float*)d_in[12];
  const float* gate_b   = (const float*)d_in[13];
  const float* cg_w1    = (const float*)d_in[14];
  const float* cg_b1    = (const float*)d_in[15];
  const float* cg_w2    = (const float*)d_in[16];
  const float* cg_b2    = (const float*)d_in[17];
  const float* sg_w     = (const float*)d_in[18];
  const float* gn_g     = (const float*)d_in[19];
  const float* gn_b     = (const float*)d_in[20];
  const float* se_w1    = (const float*)d_in[21];
  const float* se_b1    = (const float*)d_in[22];
  const float* se_w2    = (const float*)d_in[23];
  const float* se_b2    = (const float*)d_in[24];
  const float* id_scale = (const float*)d_in[25];
  const float* bias     = (const float*)d_in[26];
  float* out = (float*)d_out;

  // ---- workspace layout (~56.6 MB total) ----
  char* ws = (char*)d_ws;
  _Float16* x2h  = (_Float16*)(ws + 0);          // 16 MB
  _Float16* lfh  = (_Float16*)(ws + 16777216);   // 4 MB
  float*    outg = (float*)   (ws + 20971520);   // 32 MB
  _Float16* wcat = (_Float16*)(ws + 54525952);   // 1.44 MB
  _Float16* rch  = (_Float16*)(ws + 56000512);   // 72 KB
  _Float16* gth  = (_Float16*)(ws + 56074240);   // 32 KB
  float* p     = (float*)(ws + 56107008);
  float* g     = p + 2048;
  float* cattn = g + 2048;
  float* px2   = cattn + 2048;
  float* gf    = px2 + 2048;
  float* sin2  = gf + 512;
  float* sattn = sin2 + 32768;
  float* sums  = sattn + 65536;
  float* sumsq = sums + 2048;
  float* scl   = sumsq + 2048;
  float* shf   = scl + 2048;

  reduce_mean_f32<<<2048, 256, 0, stream>>>(x, p);
  gates_kernel<<<1, 256, 0, stream>>>(p, gc_w1, gc_w2, cg_w1, cg_b1, cg_w2, cg_b2, g, cattn);
  spatial_stats<<<256, 256, 0, stream>>>(x, g, sin2);
  sg_conv<<<256, 256, 0, stream>>>(sin2, sg_w, sattn);
  x2_kernel<<<32768, 256, 0, stream>>>(x, g, cattn, sattn, x2h);
  reduce_mean_f16<<<2048, 256, 0, stream>>>(x2h, px2);
  gf_kernel<<<1, 256, 0, stream>>>(px2, rg_w, gf);
  prep_weights<<<3088, 256, 0, stream>>>(base_w, delta_w, ksc, rc_w, gate_w, wcat, rch, gth);
  router_wmma<<<dim3(256, 16), 64, 0, stream>>>(x2h, rch, bn_g, bn_b, bn_m, bn_v, gf, lfh);
  zero_f32<<<16, 256, 0, stream>>>(sums, 4096);   // sums + sumsq contiguous
  main_wmma<<<dim3(256, 16), 256, 0, stream>>>(x2h, wcat, lfh, gth, gate_b, outg, sums, sumsq);
  gn_se<<<1, 256, 0, stream>>>(sums, sumsq, gn_g, gn_b, se_w1, se_b1, se_w2, se_b2, bias, scl, shf);
  final_kernel<<<32768, 256, 0, stream>>>(outg, x, scl, shf, id_scale, out);
  (void)in_sizes; (void)n_in; (void)out_size; (void)ws_size;
}
